// SyntaxTreeGRU_66176856097328
// MI455X (gfx1250) — compile-verified, matrix operand order checked by reading, not run on hardware
//
#include <hip/hip_runtime.h>
#include <hip/hip_bf16.h>

typedef unsigned short u16;
typedef __attribute__((ext_vector_type(16))) __bf16 v16bf;
typedef __attribute__((ext_vector_type(8)))  float  v8f;

#define HDIM   256
#define H3     768
#define LSEQ   1024
#define BATCH  64
#define NRULE  8
#define NDEPTH 10

union FragBF {
  v16bf v;
  uint4 q[2];
};

__device__ __forceinline__ u16 f2bf(float f) {
  unsigned int u = __float_as_uint(f);
  unsigned int r = (u + 0x7FFFu + ((u >> 16) & 1u)) >> 16;  // RNE to bf16
  return (u16)r;
}

__device__ __forceinline__ float sigm(float x) { return 1.0f / (1.0f + __expf(-x)); }

__device__ __forceinline__ v8f vzero() {
  v8f z;
#pragma unroll
  for (int i = 0; i < 8; ++i) z[i] = 0.0f;
  return z;
}

// ---- weight convert: in f32 [r][k][n]  ->  out bf16 [r][n][k]  (K contiguous)
__global__ void convw_k(const float* __restrict__ in, u16* __restrict__ out, int nelem) {
  int i = blockIdx.x * 256 + threadIdx.x;
  if (i >= nelem) return;
  int r   = i / (H3 * HDIM);
  int rem = i - r * (H3 * HDIM);
  int nn  = rem / HDIM;
  int k   = rem - nn * HDIM;
  out[i] = f2bf(in[(size_t)r * HDIM * H3 + (size_t)k * H3 + nn]);
}

// ---- leaves: h0[b,l,:] = tanh(emb[tok] + ntb[ntp])  (bf16)
__global__ void embed_k(const float* __restrict__ emb, const float* __restrict__ ntb,
                        const int* __restrict__ tok, const int* __restrict__ ntp,
                        u16* __restrict__ h0) {
  int bl = blockIdx.x;
  int c  = threadIdx.x;
  int t  = tok[bl];
  int p  = ntp[bl];
  float v = tanhf(emb[(size_t)t * HDIM + c] + ntb[(size_t)p * HDIM + c]);
  h0[(size_t)bl * HDIM + c] = f2bf(v);
}

// ---- per-depth binning: reset
__global__ void prep_k(int* __restrict__ order, int cap, int* __restrict__ cnt) {
  int i = blockIdx.x * 256 + threadIdx.x;
  if (i < cap) order[i] = -1;
  if (i < NRULE) cnt[i] = 0;
}

__global__ void count_k(const int* __restrict__ rule_ids, int off, int n, int* __restrict__ cnt) {
  int i = blockIdx.x * 256 + threadIdx.x;
  if (i >= BATCH * n) return;
  int b = i / n, j = i - b * n;
  atomicAdd(&cnt[rule_ids[b * (LSEQ - 1) + off + j]], 1);
}

// exclusive scan with per-rule padding to multiple of 16 (one WMMA M-tile)
__global__ void scan_k(const int* __restrict__ cnt, int* __restrict__ pstart,
                       int* __restrict__ cursor) {
  if (threadIdx.x == 0) {
    int s = 0;
    for (int r = 0; r < NRULE; ++r) {
      pstart[r] = s;
      cursor[r] = s;
      s += (cnt[r] + 15) & ~15;
    }
    pstart[NRULE] = s;
  }
}

__global__ void scatter_k(const int* __restrict__ rule_ids, int off, int n,
                          int* __restrict__ cursor, int* __restrict__ order) {
  int i = blockIdx.x * 256 + threadIdx.x;
  if (i >= BATCH * n) return;
  int b = i / n, j = i - b * n;
  int r = rule_ids[b * (LSEQ - 1) + off + j];
  int pos = atomicAdd(&cursor[r], 1);
  order[pos] = i;  // row id = b*n + j
}

// ---- 16xK (LDS, bf16 row-major) @ KxN (global, bf16 [N][K]) -> 6 16x16 f32 tiles/wave.
// Wave w owns columns w*32..w*32+31 in each gate section (s*256), so gate math is wave-local.
__device__ __forceinline__ void mm16(const u16* __restrict__ ab, const u16* __restrict__ Wt,
                                     v8f acc[2][3], int w, int hf, int l15) {
  for (int kb = 0; kb < HDIM; kb += 32) {
    // A fragment (ISA 16-bit A 16x32): lanes 0-15 K {kb..kb+7, kb+16..kb+23}, lanes 16-31 +8
    FragBF a;
    const u16* ap = ab + l15 * HDIM + kb + hf * 8;
    a.q[0] = *(const uint4*)ap;
    a.q[1] = *(const uint4*)(ap + 16);
#pragma unroll
    for (int s = 0; s < 3; ++s) {
#pragma unroll
      for (int t = 0; t < 2; ++t) {
        int ncol = s * HDIM + w * 32 + t * 16 + l15;  // column this lane supplies
        FragBF bm;
        const u16* bp = Wt + (size_t)ncol * HDIM + kb + hf * 8;
        bm.q[0] = *(const uint4*)bp;
        bm.q[1] = *(const uint4*)(bp + 16);
        acc[t][s] = __builtin_amdgcn_wmma_f32_16x16x32_bf16(
            false, a.v, false, bm.v, (short)0, acc[t][s], false, false);
      }
    }
  }
}

__global__ __launch_bounds__(256) void gemm_k(
    const u16* __restrict__ hin, u16* __restrict__ hout,
    const u16* __restrict__ WtIH, const u16* __restrict__ WtHH,
    const float* __restrict__ bih, const float* __restrict__ bhh,
    const float* __restrict__ ntb, const int* __restrict__ nt_int,
    const int* __restrict__ order, const int* __restrict__ pstart,
    int off, int n, int is_last, float* __restrict__ outf) {
  __shared__ __align__(16) u16 abuf0[16 * HDIM];  // hl, then hrt (bf16)
  __shared__ __align__(16) u16 abuf1[16 * HDIM];  // h1 (bf16)
  __shared__ int srow[16];
  __shared__ int snid[16];

  int tilebase = blockIdx.x * 16;
  if (tilebase >= pstart[NRULE]) return;  // uniform per block

  int rule = 0;
#pragma unroll
  for (int rr = 1; rr < NRULE; ++rr) rule += (tilebase >= pstart[rr]) ? 1 : 0;

  const u16* WI = WtIH + (size_t)rule * H3 * HDIM;
  const u16* WH = WtHH + (size_t)rule * H3 * HDIM;
  const float* bi = bih + rule * H3;
  const float* bh = bhh + rule * H3;

  int tid = threadIdx.x;
  if (tid < 16) {
    int sr = order[tilebase + tid];
    srow[tid] = sr;
    int nid = 0;
    if (sr >= 0) {
      int b = sr / n, j = sr - b * n;
      nid = nt_int[b * (LSEQ - 1) + off + j];
    }
    snid[tid] = nid;
  }

  int row = tid >> 4, seg = tid & 15;
  {  // cooperative load of hl tile (padded rows -> zeros)
    int sr = order[tilebase + row];
    uint4 z0 = {0, 0, 0, 0}, z1 = {0, 0, 0, 0};
    if (sr >= 0) {
      const uint4* src = (const uint4*)(hin + ((size_t)(2 * sr)) * HDIM + seg * 16);
      z0 = src[0]; z1 = src[1];
    }
    uint4* dst = (uint4*)(abuf0 + row * HDIM + seg * 16);
    dst[0] = z0; dst[1] = z1;
  }
  __syncthreads();

  int lane = tid & 31;
  int w = tid >> 5;
  int hf = lane >> 4;
  int l15 = lane & 15;

  v8f gi[2][3];
#pragma unroll
  for (int t = 0; t < 2; ++t)
#pragma unroll
    for (int s = 0; s < 3; ++s) gi[t][s] = vzero();

  mm16(abuf0, WI, gi, w, hf, l15);  // gi1 = hl @ Wih
  __syncthreads();                  // everyone done with abuf0

  // step-1 gates: h_prev = 0  =>  gh = bhh
  float h1r[2][8];
#pragma unroll
  for (int t = 0; t < 2; ++t) {
    int colH = w * 32 + t * 16 + l15;
    float bir = bi[colH], biz = bi[HDIM + colH], bin_ = bi[2 * HDIM + colH];
    float bhr = bh[colH], bhz = bh[HDIM + colH], bhn = bh[2 * HDIM + colH];
#pragma unroll
    for (int v = 0; v < 8; ++v) {
      float rg = sigm(gi[t][0][v] + bir + bhr);
      float z  = sigm(gi[t][1][v] + biz + bhz);
      float nn = tanhf(gi[t][2][v] + bin_ + rg * bhn);
      float h1 = (1.0f - z) * nn;
      h1r[t][v] = h1;
      int M = v + 8 * hf;  // C/D layout: vgpr v holds M=v (lanes 0-15) / M=v+8 (lanes 16-31)
      abuf1[M * HDIM + colH] = f2bf(h1);
    }
  }

  {  // cooperative load of hrt tile into abuf0
    int sr = order[tilebase + row];
    uint4 z0 = {0, 0, 0, 0}, z1 = {0, 0, 0, 0};
    if (sr >= 0) {
      const uint4* src = (const uint4*)(hin + ((size_t)(2 * sr + 1)) * HDIM + seg * 16);
      z0 = src[0]; z1 = src[1];
    }
    uint4* dst = (uint4*)(abuf0 + row * HDIM + seg * 16);
    dst[0] = z0; dst[1] = z1;
  }
  __syncthreads();

  v8f gh[2][3];
#pragma unroll
  for (int t = 0; t < 2; ++t)
#pragma unroll
    for (int s = 0; s < 3; ++s) { gi[t][s] = vzero(); gh[t][s] = vzero(); }

  mm16(abuf0, WI, gi, w, hf, l15);  // gi2 = hrt @ Wih
  mm16(abuf1, WH, gh, w, hf, l15);  // gh2 = h1  @ Whh

  // step-2 gates + nt_bias + store
#pragma unroll
  for (int t = 0; t < 2; ++t) {
    int colH = w * 32 + t * 16 + l15;
    float bir = bi[colH], biz = bi[HDIM + colH], bin_ = bi[2 * HDIM + colH];
    float bhr = bh[colH], bhz = bh[HDIM + colH], bhn = bh[2 * HDIM + colH];
#pragma unroll
    for (int v = 0; v < 8; ++v) {
      int M = v + 8 * hf;
      float rg = sigm(gi[t][0][v] + bir + gh[t][0][v] + bhr);
      float z  = sigm(gi[t][1][v] + biz + gh[t][1][v] + bhz);
      float nn = tanhf(gi[t][2][v] + bin_ + rg * (gh[t][2][v] + bhn));
      float h2 = (1.0f - z) * nn + z * h1r[t][v];
      int sr = srow[M];
      if (sr >= 0) {
        float o = h2 + ntb[(size_t)snid[M] * HDIM + colH];
        hout[(size_t)sr * HDIM + colH] = f2bf(o);
        if (is_last) outf[(size_t)sr * HDIM + colH] = o;  // d=9: sr==b, (B,H) fp32
      }
    }
  }
}

extern "C" void kernel_launch(void* const* d_in, const int* in_sizes, int n_in,
                              void* d_out, int out_size, void* d_ws, size_t ws_size,
                              hipStream_t stream) {
  const float* emb      = (const float*)d_in[0];
  const float* ntb      = (const float*)d_in[1];
  const float* Wih      = (const float*)d_in[2];
  const float* Whh      = (const float*)d_in[3];
  const float* bih      = (const float*)d_in[4];
  const float* bhh      = (const float*)d_in[5];
  const int*   tokens   = (const int*)d_in[6];
  const int*   nt_pre   = (const int*)d_in[7];
  const int*   nt_int   = (const int*)d_in[8];
  const int*   rule_ids = (const int*)d_in[9];

  // workspace layout (bytes): hA 32MB | hB 16MB | WtIH 3MB | WtHH 3MB | ints | order
  char* ws = (char*)d_ws;
  u16* hA     = (u16*)(ws);
  u16* hB     = (u16*)(ws + (size_t)33554432);
  u16* WtIH   = (u16*)(ws + (size_t)50331648);
  u16* WtHH   = (u16*)(ws + (size_t)53477376);
  int* cnt    = (int*)(ws + (size_t)56623104);
  int* pstart = (int*)(ws + (size_t)56623104 + 64);
  int* cursor = (int*)(ws + (size_t)56623104 + 128);
  int* order  = (int*)(ws + (size_t)56623104 + 256);

  int welem = NRULE * H3 * HDIM;
  convw_k<<<(welem + 255) / 256, 256, 0, stream>>>(Wih, WtIH, welem);
  convw_k<<<(welem + 255) / 256, 256, 0, stream>>>(Whh, WtHH, welem);
  embed_k<<<BATCH * LSEQ, HDIM, 0, stream>>>(emb, ntb, tokens, nt_pre, hA);

  int off = 0;
  for (int d = 0; d < NDEPTH; ++d) {
    int n = LSEQ >> (d + 1);
    u16* hin  = (d & 1) ? hB : hA;
    u16* hout = (d & 1) ? hA : hB;
    int rows = BATCH * n;
    int cap  = rows + 128;  // padded total <= rows + 8*15 < cap
    prep_k<<<(cap + 255) / 256, 256, 0, stream>>>(order, cap, cnt);
    count_k<<<(rows + 255) / 256, 256, 0, stream>>>(rule_ids, off, n, cnt);
    scan_k<<<1, 32, 0, stream>>>(cnt, pstart, cursor);
    scatter_k<<<(rows + 255) / 256, 256, 0, stream>>>(rule_ids, off, n, cursor, order);
    gemm_k<<<cap / 16, 256, 0, stream>>>(hin, hout, WtIH, WtHH, bih, bhh, ntb, nt_int,
                                         order, pstart, off, n,
                                         (d == NDEPTH - 1) ? 1 : 0, (float*)d_out);
    off += n;
  }
  (void)in_sizes; (void)n_in; (void)out_size; (void)ws_size;
}
